// GetAggregatedLaneEnc_25675314495809
// MI455X (gfx1250) — matrix operand order repaired and checked by reading, not compiled, not run on hardware
//
#include <hip/hip_runtime.h>
#include <hip/hip_bf16.h>

typedef __attribute__((ext_vector_type(16))) _Float16 v16h;
typedef __attribute__((ext_vector_type(8)))  float    v8f;

#define HFEAT     64      // per-tensor feature width
#define XDIM      128     // concat(lane_ht, lane_info)
#define OUTFEAT   192     // concat of all three
#define SEG_CAP   8192    // LDS score capacity per segment (stat max ~460)

// Sum across the 16-lane row group via DPP16 ROW_XMASK (pure VALU, no LDS).
// dpp_ctrl 0x160|mask = ROW_XMASK: lane reads lane^mask within its row of 16.
template <int MASK>
__device__ __forceinline__ float dpp_xor_add(float x) {
    int xi = __builtin_bit_cast(int, x);
    int yi = __builtin_amdgcn_update_dpp(xi, xi, 0x160 | MASK, 0xF, 0xF, true);
    return x + __builtin_bit_cast(float, yi);
}

__global__ __launch_bounds__(256)
void lane_aggr_kernel(const float* __restrict__ lane_ht,
                      const float* __restrict__ lane_info,
                      const float* __restrict__ lane_future,
                      const int*   __restrict__ seg,
                      const float* __restrict__ w1,   // [128,16]
                      const float* __restrict__ b1,   // [16]
                      const float* __restrict__ w2,   // [16,1]
                      const float* __restrict__ b2,   // [1]
                      float* __restrict__ out,        // [NOBS,192]
                      int M)
{
    __shared__ float s_scores[SEG_CAP];
    __shared__ float s_red[16];
    __shared__ int   s_bounds[2];

    const int tid = threadIdx.x;
    const int blk = blockIdx.x;

    // --- segment bounds via binary search on sorted seg[] ---
    if (tid < 2) {
        int target = blk + tid;            // lower_bound(blk), lower_bound(blk+1)
        int lo = 0, hi = M;
        while (lo < hi) {
            int mid = (lo + hi) >> 1;
            if (seg[mid] < target) lo = mid + 1; else hi = mid;
        }
        s_bounds[tid] = lo;
    }
    __syncthreads();
    const int start = s_bounds[0];
    int cnt = s_bounds[1] - start;
    if (cnt > SEG_CAP) cnt = SEG_CAP;      // never triggers for this distribution

    const int wave = tid >> 5;             // 8 waves / block
    const int lane = tid & 31;
    const int hi16 = lane >> 4;            // half-wave group
    const int ln   = lane & 15;            // N column (and row-writer id)

    // --- B fragments: W1 as 4x (32x16) f16 fragments, contiguous K order (matches A) ---
    // lanes 0-15 hold K=0..15 of the 32-chunk, lanes 16-31 hold K=16..31; 2 K/VGPR.
    v16h bf[4];
    #pragma unroll
    for (int f = 0; f < 4; ++f) {
        #pragma unroll
        for (int i = 0; i < 8; ++i) {
            int k0 = 32 * f + hi16 * 16 + 2 * i;
            bf[f][2*i]   = (_Float16)w1[(size_t)(k0    ) * 16 + ln];
            bf[f][2*i+1] = (_Float16)w1[(size_t)(k0 + 1) * 16 + ln];
        }
    }
    const float b1n = b1[ln];
    const float w2n = w2[ln];
    const float b20 = b2[0];

    // --- Pass 1: scores via WMMA (16 rows per tile, K=128 in 4 fragments) ---
    const int ntiles = (cnt + 15) >> 4;
    for (int t = wave; t < ntiles; t += 8) {        // wave-uniform loop
        const int row0 = start + (t << 4);
        int rowm = min(row0 + ln, M - 1);           // branch-free tail clamp

        v8f acc = {};
        #pragma unroll
        for (int f = 0; f < 4; ++f) {
            // A layout 16x32 f16: lane = row M, contiguous K = 16*hi16 + s (matches B)
            v16h af;
            #pragma unroll
            for (int i = 0; i < 8; ++i) {
                int kk = hi16 * 16 + 2 * i;
                int K  = 32 * f + kk;               // even; pair never straddles 64
                const float* src = (K < HFEAT)
                    ? (lane_ht   + (size_t)rowm * HFEAT + K)
                    : (lane_info + (size_t)rowm * HFEAT + (K - HFEAT));
                af[2*i]   = (_Float16)src[0];
                af[2*i+1] = (_Float16)src[1];
            }
            acc = __builtin_amdgcn_wmma_f32_16x16x32_f16(
                      false, af, false, bf[f], (short)0, acc, false, false);
        }

        // Layer 2: relu + dot(w2) per row; DPP ROW_XMASK butterfly over 16 lanes.
        #pragma unroll
        for (int r = 0; r < 8; ++r) {
            float h = acc[r] + b1n;
            h = h > 0.f ? h : 0.f;
            float p = h * w2n;
            p = dpp_xor_add<1>(p);
            p = dpp_xor_add<2>(p);
            p = dpp_xor_add<4>(p);
            p = dpp_xor_add<8>(p);
            if (ln == r) {                          // one writer per row
                int mrow = (t << 4) + r + 8 * hi16; // local row within segment
                if (mrow < cnt) s_scores[mrow] = p + b20;   // *1.0 soft-argmax
            }
        }
    }
    __syncthreads();

    // --- Pass 2a: block max ---
    float mx = -3.402823466e38f;
    for (int i = tid; i < cnt; i += 256) mx = fmaxf(mx, s_scores[i]);
    #pragma unroll
    for (int o = 16; o > 0; o >>= 1) mx = fmaxf(mx, __shfl_xor(mx, o, 32));
    if (lane == 0) s_red[wave] = mx;
    __syncthreads();
    if (tid == 0) {
        float m2 = s_red[0];
        #pragma unroll
        for (int w = 1; w < 8; ++w) m2 = fmaxf(m2, s_red[w]);
        s_red[8] = m2;
    }
    __syncthreads();
    mx = s_red[8];

    // --- Pass 2b: exp + block sum (in-place ex in LDS) ---
    float sm = 0.f;
    for (int i = tid; i < cnt; i += 256) {
        float e = __expf(s_scores[i] - mx);
        s_scores[i] = e;
        sm += e;
    }
    #pragma unroll
    for (int o = 16; o > 0; o >>= 1) sm += __shfl_xor(sm, o, 32);
    if (lane == 0) s_red[wave] = sm;
    __syncthreads();
    if (tid == 0) {
        float s2 = 0.f;
        #pragma unroll
        for (int w = 0; w < 8; ++w) s2 += s_red[w];
        s_red[9] = (s2 > 0.f) ? (1.0f / s2) : 0.f;
    }
    __syncthreads();
    const float inv = s_red[9];

    // --- Pass 3: weighted pooling, one feature column per thread (coalesced).
    // lane_ht / lane_info were just read by pass 1 => L2-resident (192MB L2).
    // lane_future is single-use => non-temporal so it doesn't evict them.
    if (tid < OUTFEAT) {
        float a = 0.f;
        if (tid < 2 * HFEAT) {
            const float* src = (tid < HFEAT) ? lane_ht : lane_info;
            const int    off = tid & (HFEAT - 1);
            #pragma unroll 4
            for (int i = 0; i < cnt; ++i)
                a += src[(size_t)(start + i) * HFEAT + off] * s_scores[i];
        } else {
            const int off = tid - 2 * HFEAT;
            #pragma unroll 4
            for (int i = 0; i < cnt; ++i) {
                float v = __builtin_nontemporal_load(
                    lane_future + (size_t)(start + i) * HFEAT + off);
                a += v * s_scores[i];
            }
        }
        out[(size_t)blk * OUTFEAT + tid] = a * inv;
    }
}

extern "C" void kernel_launch(void* const* d_in, const int* in_sizes, int n_in,
                              void* d_out, int out_size, void* d_ws, size_t ws_size,
                              hipStream_t stream) {
    const float* lane_ht     = (const float*)d_in[0];
    const float* lane_info   = (const float*)d_in[1];
    const float* lane_future = (const float*)d_in[2];
    const int*   seg         = (const int*)  d_in[3];
    const float* w1          = (const float*)d_in[4];
    const float* b1          = (const float*)d_in[5];
    const float* w2          = (const float*)d_in[6];
    const float* b2          = (const float*)d_in[7];
    float* out = (float*)d_out;

    const int M    = in_sizes[3];            // same_obstacle_mask is [M,1]
    const int nobs = out_size / OUTFEAT;     // 1024

    lane_aggr_kernel<<<nobs, 256, 0, stream>>>(lane_ht, lane_info, lane_future,
                                               seg, w1, b1, w2, b2, out, M);
}
